// LDRFat_23476291240731
// MI455X (gfx1250) — matrix operand order repaired and checked by reading, not compiled, not run on hardware
//
#include <hip/hip_runtime.h>
#include <hip/hip_bf16.h>

// ---------------------------------------------------------------------------
// LDRFat MLP forward for MI455X (gfx1250, wave32, WMMA).
//   h   = x @ W                      [16384,3072]x[3072,3072]
//   h   = relu(h @ fc_w^T + fc_b)    [16384,3072]x[3072,512]
//   out = h @ logits_w^T + logits_b  [16384,512]x[512,10]
// bf16 WMMA (f32 accumulate); Xbf + W^T live in the 192MB L2.
// Block tile 256x128x32, 8 waves (4x2), wave tile 64x64 ->
// 16 x v_wmma_f32_16x16x32_bf16 per k-step per wave.
// Global->LDS staging uses gfx1250 ASYNC loads (ASYNCcnt), triple-buffered
// LDS, prefetch distance 2; pipeline rotation unrolled 3x (KT % 3 == 0).
// ---------------------------------------------------------------------------

#define BDIM 16384
#define NDIM 3072
#define FCDIM 512
#define CLSDIM 10

typedef __attribute__((ext_vector_type(16))) __bf16 v16bf;
typedef __attribute__((ext_vector_type(8)))  __bf16 v8bf;
typedef __attribute__((ext_vector_type(8)))  float  v8f;
typedef __attribute__((ext_vector_type(4)))  float  v4f;
typedef int v4i_t __attribute__((vector_size(16)));   // matches builtin proto

union BF16x16 { v16bf v; v8bf h[2]; };

// ---- gfx1250 async global->LDS copy + ASYNCcnt wait (builtin or asm) ------
__device__ __forceinline__ void async_ld_b128(const __bf16* g, __bf16* l) {
#if __has_builtin(__builtin_amdgcn_global_load_async_to_lds_b128)
    __builtin_amdgcn_global_load_async_to_lds_b128(
        (__attribute__((address_space(1))) v4i_t*)(v4i_t*)(void*)g,
        (__attribute__((address_space(3))) v4i_t*)(v4i_t*)(void*)l,
        /*offset=*/0, /*cpol=*/0);
#else
    unsigned loff = (unsigned)(unsigned long long)
        (__attribute__((address_space(3))) char*)(void*)l;
    unsigned long long ga = (unsigned long long)g;
    asm volatile("global_load_async_to_lds_b128 %0, %1, off"
                 :: "v"(loff), "v"(ga) : "memory");
#endif
}

#if __has_builtin(__builtin_amdgcn_s_wait_asynccnt)
#define WAIT_ASYNC(n) __builtin_amdgcn_s_wait_asynccnt(n)
#else
#define WAIT_ASYNC(n) asm volatile("s_wait_asynccnt %0" :: "i"(n) : "memory")
#endif

// ---------------------------------------------------------------------------
// fp32 -> bf16 elementwise convert (vectorized x4). n must be multiple of 4.
// ---------------------------------------------------------------------------
__global__ __launch_bounds__(256) void cvt_f32_bf16(const float* __restrict__ src,
                                                    __bf16* __restrict__ dst,
                                                    long long n4) {
    long long i = (long long)blockIdx.x * blockDim.x + threadIdx.x;
    if (i >= n4) return;
    v4f f = ((const v4f*)src)[i];
    __bf16 b[4];
    b[0] = (__bf16)f.x; b[1] = (__bf16)f.y; b[2] = (__bf16)f.z; b[3] = (__bf16)f.w;
    ((ulong1*)dst)[i] = *(ulong1*)b;
}

// ---------------------------------------------------------------------------
// Transpose + convert: dst[n][k] = (bf16)src[k][n], square dim x dim.
// ---------------------------------------------------------------------------
__global__ __launch_bounds__(256) void transpose_cvt(const float* __restrict__ src,
                                                     __bf16* __restrict__ dst,
                                                     int dim) {
    __shared__ float t[32][33];
    const int bx = blockIdx.x * 32;
    const int by = blockIdx.y * 32;
    const int tx = threadIdx.x;       // 0..31
    const int ty = threadIdx.y;       // 0..7
#pragma unroll
    for (int j = 0; j < 32; j += 8)
        t[ty + j][tx] = src[(size_t)(by + ty + j) * dim + (bx + tx)];
    __syncthreads();
#pragma unroll
    for (int j = 0; j < 32; j += 8)
        dst[(size_t)(bx + ty + j) * dim + (by + tx)] = (__bf16)t[tx][ty + j];
}

// ---------------------------------------------------------------------------
// bf16 WMMA GEMM:  C[M,Nc] = A[M,K] * Bt[Nc,K]^T   (Bt is k-contiguous)
// M multiple of 256, Nc multiple of 128, K multiple of 96 (KT % 3 == 0).
// EPI==1: add bias[col] + ReLU before bf16 store.
// ---------------------------------------------------------------------------
template <int EPI>
__global__ __launch_bounds__(256) void gemm_bf16(const __bf16* __restrict__ A,
                                                 const __bf16* __restrict__ Bt,
                                                 __bf16* __restrict__ C,
                                                 const float* __restrict__ bias,
                                                 int M, int Nc, int K) {
    __shared__ __bf16 As[3][256][32];   // 3 x 16 KB
    __shared__ __bf16 Bs[3][128][32];   // 3 x  8 KB

    const int tid  = threadIdx.x;          // 0..255
    const int lane = tid & 31;
    const int wave = tid >> 5;             // 0..7
    const int wm0  = (wave >> 1) * 64;     // wave row origin in block tile
    const int wn0  = (wave & 1) * 64;      // wave col origin in block tile
    const int lr   = lane & 15;            // A: row sel / B: col sel
    const int lh   = lane >> 4;            // K-half selector

    // staging: A = one 32-elem row per thread (4x16B), B = half-row (2x16B)
    const int brow  = tid >> 1;            // 0..127
    const int bhalf = (tid & 1) * 16;      // 0 or 16

    const __bf16* gA = A  + (size_t)(blockIdx.y * 256 + tid) * K;
    const __bf16* gB = Bt + (size_t)(blockIdx.x * 128 + brow) * K + bhalf;

    v8f acc[4][4];
#pragma unroll
    for (int t = 0; t < 4; ++t)
#pragma unroll
        for (int u = 0; u < 4; ++u) acc[t][u] = (v8f)0.f;

    // issue 6 async global->LDS ops for k-block kb into LDS buffer `buf`
    auto stage = [&](int kb, int buf) {
        const __bf16* pA = gA + (size_t)kb * 32;
        const __bf16* pB = gB + (size_t)kb * 32;
#pragma unroll
        for (int j = 0; j < 4; ++j)
            async_ld_b128(pA + j * 8, &As[buf][tid][j * 8]);
        async_ld_b128(pB,     &Bs[buf][brow][bhalf]);
        async_ld_b128(pB + 8, &Bs[buf][brow][bhalf + 8]);
    };

    // fragment loads (ISA 7.12.2 layouts) + 16 WMMAs from LDS buffer `cb`
    auto compute = [&](int cb) {
        BF16x16 af[4], bf[4];
#pragma unroll
        for (int t = 0; t < 4; ++t) {
            const __bf16* ap = &As[cb][wm0 + t * 16 + lr][lh * 8];
            af[t].h[0] = *(const v8bf*)ap;
            af[t].h[1] = *(const v8bf*)(ap + 16);
        }
#pragma unroll
        for (int u = 0; u < 4; ++u) {
            const __bf16* bp = &Bs[cb][wn0 + u * 16 + lr][lh * 8];
            bf[u].h[0] = *(const v8bf*)bp;
            bf[u].h[1] = *(const v8bf*)(bp + 16);
        }
#pragma unroll
        for (int t = 0; t < 4; ++t)
#pragma unroll
            for (int u = 0; u < 4; ++u)
                acc[t][u] = __builtin_amdgcn_wmma_f32_16x16x32_bf16(
                    false, af[t].v, false, bf[u].v,
                    (short)0, acc[t][u], false, false);
    };

    // one pipeline step: compute buffer `buf` for k-block kb,
    // prefetch k-block kb+2 into the buffer read by the previous step.
    auto step = [&](int kb, int buf, int pbuf, int KT) {
        const bool pf = (kb + 2) < KT;
        if (pf) stage(kb + 2, pbuf);
        compute(buf);
        if (pf) WAIT_ASYNC(6);             // stage kb+1 landed, kb+2 in flight
        else    WAIT_ASYNC(0);             // drain tail
        __syncthreads();
    };

    const int KT = K >> 5;                 // number of 32-wide k-blocks, %3==0

    // ---- prologue: stages 0 and 1 in flight; wait for stage 0 ----
    stage(0, 0);
    stage(1, 1);
    WAIT_ASYNC(6);                         // stage 0 landed (stage 1 in flight)
    __syncthreads();

    // ---- steady-state: rotation unrolled 3x, constant buffer indices ----
    for (int kb = 0; kb < KT; kb += 3) {
        step(kb + 0, 0, 2, KT);
        step(kb + 1, 1, 0, KT);
        step(kb + 2, 2, 1, KT);
    }

    // ---- epilogue: D layout — VGPR r: row = 8*lh + r, col = lr ----
#pragma unroll
    for (int t = 0; t < 4; ++t) {
#pragma unroll
        for (int u = 0; u < 4; ++u) {
            const int row0 = blockIdx.y * 256 + wm0 + t * 16 + lh * 8;
            const int col  = blockIdx.x * 128 + wn0 + u * 16 + lr;
            const float bv = EPI ? bias[col] : 0.f;
            v8f c = acc[t][u];
#pragma unroll
            for (int r = 0; r < 8; ++r) {
                float v = c[r];
                if (EPI) v = fmaxf(v + bv, 0.f);
                C[(size_t)(row0 + r) * Nc + col] = (__bf16)v;
            }
        }
    }
}

// ---------------------------------------------------------------------------
// Head: out[b][c] = sum_k H2[b][k]*lw[c][k] + lb[c]   (K=512, CLS=10)
// ---------------------------------------------------------------------------
__global__ __launch_bounds__(256) void head_kernel(const __bf16* __restrict__ H2,
                                                   const float* __restrict__ lw,
                                                   const float* __restrict__ lb,
                                                   float* __restrict__ out,
                                                   int M, int K, int C) {
    long long idx = (long long)blockIdx.x * blockDim.x + threadIdx.x;
    int b = (int)(idx >> 4);      // 16 lanes per row (C padded to 16)
    int c = (int)(idx & 15);
    if (b >= M || c >= C) return;
    const __bf16* hp = H2 + (size_t)b * K;
    const float*  wp = lw + (size_t)c * K;
    float s = 0.f;
#pragma unroll 8
    for (int k = 0; k < K; ++k)
        s += (float)hp[k] * wp[k];
    out[(size_t)b * C + c] = s + lb[c];
}

// ---------------------------------------------------------------------------
extern "C" void kernel_launch(void* const* d_in, const int* in_sizes, int n_in,
                              void* d_out, int out_size, void* d_ws, size_t ws_size,
                              hipStream_t stream) {
    (void)in_sizes; (void)n_in; (void)out_size; (void)ws_size;
    const float* x    = (const float*)d_in[0];   // [B,N]
    const float* W    = (const float*)d_in[1];   // [N,N]
    const float* fc_w = (const float*)d_in[2];   // [FC,N]
    const float* fc_b = (const float*)d_in[3];   // [FC]
    const float* lw   = (const float*)d_in[4];   // [CLS,FC]
    const float* lb   = (const float*)d_in[5];   // [CLS]
    float* out = (float*)d_out;                  // [B,CLS]

    char* ws = (char*)d_ws;
    size_t off = 0;
    auto carve = [&](size_t bytes) {
        void* p = ws + off;
        off = (off + bytes + 255) & ~(size_t)255;
        return p;
    };
    __bf16* Xbf  = (__bf16*)carve((size_t)BDIM * NDIM * 2);   // 100.7 MB
    __bf16* WT   = (__bf16*)carve((size_t)NDIM * NDIM * 2);   //  18.9 MB
    __bf16* FCbf = (__bf16*)carve((size_t)FCDIM * NDIM * 2);  //   3.1 MB
    __bf16* Hbf  = (__bf16*)carve((size_t)BDIM * NDIM * 2);   // 100.7 MB
    __bf16* H2bf = (__bf16*)carve((size_t)BDIM * FCDIM * 2);  //  16.8 MB

    // 1) down-convert activations & weights (Xbf+WT then live in 192MB L2)
    {
        long long n4 = (long long)BDIM * NDIM / 4;
        cvt_f32_bf16<<<(unsigned)((n4 + 255) / 256), 256, 0, stream>>>(x, Xbf, n4);
    }
    {
        dim3 g(NDIM / 32, NDIM / 32), blk(32, 8);
        transpose_cvt<<<g, blk, 0, stream>>>(W, WT, NDIM);
    }
    {
        long long n4 = (long long)FCDIM * NDIM / 4;
        cvt_f32_bf16<<<(unsigned)((n4 + 255) / 256), 256, 0, stream>>>(fc_w, FCbf, n4);
    }

    // 2) GEMM1: Hbf = Xbf @ WT^T      (i.e. x @ W)
    {
        dim3 g(NDIM / 128, BDIM / 256);
        gemm_bf16<0><<<g, 256, 0, stream>>>(Xbf, WT, Hbf, nullptr, BDIM, NDIM, NDIM);
    }
    // 3) GEMM2 + bias + ReLU: H2bf = relu(Hbf @ fc_w^T + fc_b)
    {
        dim3 g(FCDIM / 128, BDIM / 256);
        gemm_bf16<1><<<g, 256, 0, stream>>>(Hbf, FCbf, H2bf, fc_b, BDIM, FCDIM, NDIM);
    }
    // 4) head
    {
        long long thr = (long long)BDIM * 16;
        head_kernel<<<(unsigned)((thr + 255) / 256), 256, 0, stream>>>(
            H2bf, lw, lb, out, BDIM, FCDIM, CLSDIM);
    }
}
